// NTM_39711267619345
// MI455X (gfx1250) — matrix operand order repaired and checked by reading, not compiled, not run on hardware
//
#include <hip/hip_runtime.h>
#include <math.h>

// ---------------------------------------------------------------------------
// NTM on MI455X (gfx1250, wave32):
//   K1: WMMA bf16 GEMM  Gin = Emb[x] @ W_ih[:, :256]^T + b_ih      (512x4096x256)
//   K2: persistent grid-synced recurrence (serial part only), L2-resident W_hh,
//       writes HR[t] = [h_t (1024) | read_t (64)] in bf16
//   K3: WMMA bf16 GEMM  logits = HR @ Wo^T + bo                    (512x50257x1088)
//       M is the fast grid axis -> Wo streamed from HBM exactly once (L2 reuse),
//       64x32 wave tiles -> 2 cvt_pk + 32B of loads per WMMA.
// ---------------------------------------------------------------------------

typedef __attribute__((ext_vector_type(16))) __bf16 v16bf;
typedef __attribute__((ext_vector_type(8)))  float  v8f;

#define NV   50257
#define ND   256
#define NH   1024
#define NMU  1024
#define NMD  64
#define NS   512
#define NKO  1088      // H + MD
#define NWGR 32        // workgroups in recurrence kernel

__device__ __forceinline__ float sigm(float x)      { return 1.f / (1.f + __expf(-x)); }
__device__ __forceinline__ float softplusf(float x) { return (x > 20.f) ? x : log1pf(__expf(x)); }

// ---------------- WMMA fragment loaders (ISA 7.12.2 layouts, NT GEMM) -------

// A-matrix 16x32 bf16: lane m=lane&15, half=lane>>4.
// VGPR 0-3 hold K = 8*half + {0..7}; VGPR 4-7 hold K = 16 + 8*half + {0..7}.
__device__ __forceinline__ v16bf frag_a_bf16(const __bf16* rowp, int kk) {
  const int lane = threadIdx.x & 31;
  const int half = lane >> 4;
  union U { uint4 q; __bf16 h[8]; } u0, u1;
  u0.q = *(const uint4*)(rowp + kk + 8 * half);
  u1.q = *(const uint4*)(rowp + kk + 16 + 8 * half);
  v16bf f;
#pragma unroll
  for (int i = 0; i < 8; ++i) { f[i] = u0.h[i]; f[8 + i] = u1.h[i]; }
  return f;
}

// Same A layout but source is fp32 (convert in-register to bf16).
__device__ __forceinline__ v16bf frag_a_f32(const float* rowp, int kk) {
  const int lane = threadIdx.x & 31;
  const int half = lane >> 4;
  const float4* p0 = (const float4*)(rowp + kk + 8 * half);
  const float4* p1 = (const float4*)(rowp + kk + 16 + 8 * half);
  float4 a0 = p0[0], a1 = p0[1], b0 = p1[0], b1 = p1[1];
  v16bf f;
  f[0]  = (__bf16)a0.x; f[1]  = (__bf16)a0.y; f[2]  = (__bf16)a0.z; f[3]  = (__bf16)a0.w;
  f[4]  = (__bf16)a1.x; f[5]  = (__bf16)a1.y; f[6]  = (__bf16)a1.z; f[7]  = (__bf16)a1.w;
  f[8]  = (__bf16)b0.x; f[9]  = (__bf16)b0.y; f[10] = (__bf16)b0.z; f[11] = (__bf16)b0.w;
  f[12] = (__bf16)b1.x; f[13] = (__bf16)b1.y; f[14] = (__bf16)b1.z; f[15] = (__bf16)b1.w;
  return f;
}

// B-matrix 32x16 bf16: lane n=lane&15, halfK=lane>>4; elements 0-7 hold
// K = 8*halfK + {0..7}, elements 8-15 hold K = 16 + 8*halfK + {0..7}, from fp32 row (NT layout: B column = Wo row).
__device__ __forceinline__ v16bf frag_b_f32(const float* rowp, int kk) {
  const int lane = threadIdx.x & 31;
  const int half = lane >> 4;
  const float4* p0 = (const float4*)(rowp + kk + 8 * half);
  const float4* p1 = (const float4*)(rowp + kk + 16 + 8 * half);
  float4 q0 = p0[0], q1 = p0[1], q2 = p1[0], q3 = p1[1];
  v16bf f;
  f[0]  = (__bf16)q0.x; f[1]  = (__bf16)q0.y; f[2]  = (__bf16)q0.z; f[3]  = (__bf16)q0.w;
  f[4]  = (__bf16)q1.x; f[5]  = (__bf16)q1.y; f[6]  = (__bf16)q1.z; f[7]  = (__bf16)q1.w;
  f[8]  = (__bf16)q2.x; f[9]  = (__bf16)q2.y; f[10] = (__bf16)q2.z; f[11] = (__bf16)q2.w;
  f[12] = (__bf16)q3.x; f[13] = (__bf16)q3.y; f[14] = (__bf16)q3.z; f[15] = (__bf16)q3.w;
  return f;
}

// ---------------- K1: Gin = Emb[x] @ W_ih[:, :256]^T + b_ih -----------------
// Block: 256 thr = 8 waves (2M x 4N); block tile 32M x 256N; wave tile 16x64.
// grid.x = M tiles (fast axis), grid.y = N tiles.
__global__ __launch_bounds__(256) void ntm_gin_gemm(
    const int* __restrict__ x, const float* __restrict__ emb,
    const float* __restrict__ W_ih, const float* __restrict__ b_ih,
    float* __restrict__ Gin) {
  const int wave = threadIdx.x >> 5;
  const int wm = wave >> 2, wn = wave & 3;
  const int lane = threadIdx.x & 31;
  const int tileM = blockIdx.x * 32 + wm * 16;
  const int tileN = blockIdx.y * 256 + wn * 64;
  const float* arow = emb + (size_t)x[tileM + (lane & 15)] * ND;
  const float* brow[4];
#pragma unroll
  for (int s = 0; s < 4; ++s)
    brow[s] = W_ih + (size_t)(tileN + s * 16 + (lane & 15)) * (ND + NMD);
  v8f acc[4] = {};
  for (int kk = 0; kk < ND; kk += 32) {
    v16bf a = frag_a_f32(arow, kk);
#pragma unroll
    for (int s = 0; s < 4; ++s) {
      v16bf b = frag_b_f32(brow[s], kk);
      acc[s] = __builtin_amdgcn_wmma_f32_16x16x32_bf16(
          false, a, false, b, (short)0, acc[s], false, false);
    }
  }
  const int halfC = lane >> 4, nc = lane & 15;
#pragma unroll
  for (int s = 0; s < 4; ++s) {
    int v = tileN + s * 16 + nc;
    float bias = b_ih[v];
#pragma unroll
    for (int r = 0; r < 8; ++r) {
      int m = tileM + r + 8 * halfC;
      Gin[(size_t)m * 4096 + v] = acc[s][r] + bias;
    }
  }
}

// ---------------- K3: logits = HR @ Wo^T + bo -------------------------------
// Block: 256 thr = 8 waves as (2 M-waves x 4 N-waves); block tile 128M x 128N.
// Wave tile 64M x 32N = 4 M-subtiles x 2 N-subtiles (8 accumulators).
// grid.x = M blocks (fast axis, 4 total) so concurrent blocks share the same
// Wo slice via L2 -> Wo (218 MB fp32) crosses HBM once.
__global__ __launch_bounds__(256) void ntm_out_gemm(
    const __bf16* __restrict__ HR, const float* __restrict__ Wo,
    const float* __restrict__ bo, float* __restrict__ out) {
  const int wave = threadIdx.x >> 5;
  const int wm = wave >> 2, wn = wave & 3;        // wm: 0..1, wn: 0..3
  const int lane = threadIdx.x & 31;
  const int tileM = blockIdx.x * 128 + wm * 64;   // + i*16, i=0..3
  const int tileN = blockIdx.y * 128 + wn * 32;   // + s*16, s=0..1

  const __bf16* arow0 = HR + (size_t)(tileM + (lane & 15)) * NKO;
  const float* brow[2];
#pragma unroll
  for (int s = 0; s < 2; ++s) {
    int v = tileN + s * 16 + (lane & 15);
    if (v >= NV) v = NV - 1;                      // clamp loads; stores guarded
    brow[s] = Wo + (size_t)v * NKO;
  }

  v8f acc[4][2] = {};
  for (int kk = 0; kk < NKO; kk += 32) {
    v16bf bfr[2];
#pragma unroll
    for (int s = 0; s < 2; ++s) {
      if (kk + 32 < NKO) __builtin_prefetch(brow[s] + kk + 32, 0, 0);
      bfr[s] = frag_b_f32(brow[s], kk);
    }
#pragma unroll
    for (int i = 0; i < 4; ++i) {
      v16bf a = frag_a_bf16(arow0 + (size_t)i * 16 * NKO, kk);
#pragma unroll
      for (int s = 0; s < 2; ++s) {
        acc[i][s] = __builtin_amdgcn_wmma_f32_16x16x32_bf16(
            false, a, false, bfr[s], (short)0, acc[i][s], false, false);
      }
    }
  }

  const int halfC = lane >> 4, nc = lane & 15;
#pragma unroll
  for (int s = 0; s < 2; ++s) {
    int v = tileN + s * 16 + nc;
    if (v >= NV) continue;
    float bias = bo[v];
#pragma unroll
    for (int i = 0; i < 4; ++i) {
#pragma unroll
      for (int r = 0; r < 8; ++r) {
        int m = tileM + i * 16 + r + 8 * halfC;
        out[(size_t)m * NV + v] = acc[i][s][r] + bias;
      }
    }
  }
}

// ---------------- grid-wide sense barrier for the recurrence ----------------
__device__ __forceinline__ void grid_sync(unsigned* cnt, unsigned* gen) {
  __syncthreads();
  if (threadIdx.x == 0) {
    __threadfence();
    unsigned g = __hip_atomic_load(gen, __ATOMIC_RELAXED, __HIP_MEMORY_SCOPE_AGENT);
    unsigned p = __hip_atomic_fetch_add(cnt, 1u, __ATOMIC_ACQ_REL, __HIP_MEMORY_SCOPE_AGENT);
    if (p == (unsigned)NWGR - 1) {
      __hip_atomic_store(cnt, 0u, __ATOMIC_RELAXED, __HIP_MEMORY_SCOPE_AGENT);
      __hip_atomic_fetch_add(gen, 1u, __ATOMIC_RELEASE, __HIP_MEMORY_SCOPE_AGENT);
    } else {
      while (__hip_atomic_load(gen, __ATOMIC_ACQUIRE, __HIP_MEMORY_SCOPE_AGENT) == g)
        __builtin_amdgcn_s_sleep(1);
    }
    __threadfence();
  }
  __syncthreads();
}

// ---------------- K2: the serial NTM recurrence -----------------------------
// 32 WGs x 256 threads, 4 grid barriers per step.
__global__ __launch_bounds__(256) void ntm_recur(
    const float* __restrict__ W_ih, const float* __restrict__ W_hh,
    const float* __restrict__ b_hh,
    const float* __restrict__ Wk, const float* __restrict__ bk,
    const float* __restrict__ Wb, const float* __restrict__ bb,
    const float* __restrict__ We, const float* __restrict__ be,
    const float* __restrict__ Wa, const float* __restrict__ ba,
    const float* __restrict__ Gin, __bf16* __restrict__ hr,
    float* __restrict__ memb,   // [2][MU*MD]
    float* __restrict__ hbuf,   // [2][H]
    float* __restrict__ cbuf,   // [H]
    float* __restrict__ readv,  // [MD]
    float* __restrict__ keyv, float* __restrict__ erasev, float* __restrict__ addv,
    float* __restrict__ betav, float* __restrict__ simv,
    unsigned* __restrict__ sync) {
  __shared__ float lds_w[NMU];
  __shared__ float red[256];
  const int w = blockIdx.x, tid = threadIdx.x;
  const int lane = tid & 31, wave = tid >> 5;

  for (int t = 0; t < NS; ++t) {
    const float* hprev = hbuf + (t & 1) * NH;
    float* hnext = hbuf + ((t + 1) & 1) * NH;
    const float* memold = memb + (t & 1) * (NMU * NMD);
    float* memnew = memb + ((t + 1) & 1) * (NMU * NMD);

    // ---- phase 1: gates = Gin + W_hh@h + W_ih[:,256:]@read ; LSTM cell ----
    {
      const int j = w * 32 + (tid >> 3);   // h index handled by this 8-thread group
      const int sub = tid & 7;
      float g4[4];
#pragma unroll
      for (int g = 0; g < 4; ++g) {
        const int row = j + g * NH;
        const float* wh = W_hh + (size_t)row * NH;
        const float* wi = W_ih + (size_t)row * (ND + NMD) + ND;
        float acc = 0.f;
        const int k0 = sub * 128;
        for (int k = k0; k < k0 + 128; ++k) acc += wh[k] * hprev[k];
        const int r0 = sub * 8;
        for (int d = r0; d < r0 + 8; ++d) acc += wi[d] * readv[d];
        for (int o = 4; o; o >>= 1) acc += __shfl_xor(acc, o, 8);
        g4[g] = acc;
      }
      if (sub == 0) {
        const float* gi_t = Gin + (size_t)t * 4096;
        float gi = g4[0] + gi_t[j]          + b_hh[j];
        float gf = g4[1] + gi_t[j + 1024]   + b_hh[j + 1024];
        float gg = g4[2] + gi_t[j + 2048]   + b_hh[j + 2048];
        float go = g4[3] + gi_t[j + 3072]   + b_hh[j + 3072];
        float c2 = sigm(gf) * cbuf[j] + sigm(gi) * tanhf(gg);
        float h2 = sigm(go) * tanhf(c2);
        cbuf[j] = c2;
        hnext[j] = h2;
        hr[(size_t)t * NKO + j] = (__bf16)h2;
      }
    }
    grid_sync(sync, sync + 1);

    // ---- phase 2: key/erase/add/beta GEMVs (193 rows over 256 waves) ------
    {
      const int gw = w * 8 + wave;
      if (gw < 193) {
        const float* wr; float bias; int kind, idx;
        if (gw < 64)       { wr = Wk + (size_t)gw * NH;         bias = bk[gw];       kind = 0; idx = gw; }
        else if (gw < 128) { wr = We + (size_t)(gw - 64) * NH;  bias = be[gw - 64];  kind = 1; idx = gw - 64; }
        else if (gw < 192) { wr = Wa + (size_t)(gw - 128) * NH; bias = ba[gw - 128]; kind = 2; idx = gw - 128; }
        else               { wr = Wb;                           bias = bb[0];        kind = 3; idx = 0; }
        float acc = 0.f;
        for (int k = lane; k < NH; k += 32) acc += wr[k] * hnext[k];
        for (int o = 16; o; o >>= 1) acc += __shfl_xor(acc, o, 32);
        if (lane == 0) {
          float v = acc + bias;
          if (kind == 0)      keyv[idx]   = v;
          else if (kind == 1) erasev[idx] = sigm(v);
          else if (kind == 2) addv[idx]   = tanhf(v);
          else                betav[0]    = softplusf(v);
        }
      }
    }
    grid_sync(sync, sync + 1);

    // ---- phase 3: sim[m] = beta * cos(key, mem[m])  (32 rows / WG) --------
    {
      float kd  = keyv[lane];
      float kd2 = keyv[lane + 32];
      float kn = kd * kd + kd2 * kd2;
      for (int o = 16; o; o >>= 1) kn += __shfl_xor(kn, o, 32);
      float keynorm = fmaxf(sqrtf(kn), 1e-12f);
      float beta = betav[0];
#pragma unroll
      for (int r = 0; r < 4; ++r) {
        const int m = w * 32 + wave * 4 + r;
        const float* mr = memold + (size_t)m * NMD;
        float a = mr[lane], a2 = mr[lane + 32];
        float dot = a * kd + a2 * kd2;
        float n2  = a * a + a2 * a2;
        for (int o = 16; o; o >>= 1) { dot += __shfl_xor(dot, o, 32); n2 += __shfl_xor(n2, o, 32); }
        if (lane == 0) {
          float mn = fmaxf(sqrtf(n2), 1e-12f);
          simv[m] = beta * dot / (keynorm * mn);
        }
      }
    }
    grid_sync(sync, sync + 1);

    // ---- phase 4: softmax -> w ; read = w@mem ; mem update ----------------
    {
      float sv[4]; float lmax = -INFINITY;
#pragma unroll
      for (int r = 0; r < 4; ++r) { sv[r] = simv[tid + r * 256]; lmax = fmaxf(lmax, sv[r]); }
      red[tid] = lmax; __syncthreads();
      for (int o = 128; o; o >>= 1) { if (tid < o) red[tid] = fmaxf(red[tid], red[tid + o]); __syncthreads(); }
      float gmax = red[0]; __syncthreads();
      float lsum = 0.f;
#pragma unroll
      for (int r = 0; r < 4; ++r) { float e = __expf(sv[r] - gmax); lds_w[tid + r * 256] = e; lsum += e; }
      red[tid] = lsum; __syncthreads();
      for (int o = 128; o; o >>= 1) { if (tid < o) red[tid] += red[tid + o]; __syncthreads(); }
      float inv = 1.f / red[0]; __syncthreads();
#pragma unroll
      for (int r = 0; r < 4; ++r) lds_w[tid + r * 256] *= inv;
      __syncthreads();

      // read: WG w owns columns 2w, 2w+1
#pragma unroll
      for (int cc = 0; cc < 2; ++cc) {
        const int d = w * 2 + cc;
        float s = 0.f;
        for (int m = tid; m < NMU; m += 256) s += lds_w[m] * memold[(size_t)m * NMD + d];
        red[tid] = s; __syncthreads();
        for (int o = 128; o; o >>= 1) { if (tid < o) red[tid] += red[tid + o]; __syncthreads(); }
        if (tid == 0) {
          float rv = red[0];
          readv[d] = rv;
          hr[(size_t)t * NKO + NH + d] = (__bf16)rv;
        }
        __syncthreads();
      }

      // mem update: rows [w*32, w*32+32)
      for (int e = tid; e < 32 * NMD; e += 256) {
        const int m = w * 32 + (e >> 6);
        const int d = e & 63;
        float wm = lds_w[m];
        memnew[(size_t)m * NMD + d] =
            memold[(size_t)m * NMD + d] * (1.f - wm * erasev[d]) + wm * addv[d];
      }
    }
    grid_sync(sync, sync + 1);
  }
}

// ---------------- init: state zeroing + memory_init copy --------------------
__global__ void ntm_init(const float* __restrict__ mem0, float* __restrict__ memb,
                         float* __restrict__ hbuf, float* __restrict__ cbuf,
                         float* __restrict__ readv, unsigned* __restrict__ sync) {
  const int i = blockIdx.x * 256 + threadIdx.x;
  if (i < NMU * NMD) memb[i] = mem0[i];
  if (i < 2 * NH)    hbuf[i] = 0.f;
  if (i < NH)        cbuf[i] = 0.f;
  if (i < NMD)       readv[i] = 0.f;
  if (i < 2)         sync[i] = 0u;
}

// ---------------------------------------------------------------------------
extern "C" void kernel_launch(void* const* d_in, const int* in_sizes, int n_in,
                              void* d_out, int out_size, void* d_ws, size_t ws_size,
                              hipStream_t stream) {
  const int*   x    = (const int*)  d_in[0];
  const float* emb  = (const float*)d_in[1];
  const float* W_ih = (const float*)d_in[2];
  const float* b_ih = (const float*)d_in[3];
  const float* W_hh = (const float*)d_in[4];
  const float* b_hh = (const float*)d_in[5];
  const float* Wk   = (const float*)d_in[6];
  const float* bk   = (const float*)d_in[7];
  const float* Wb   = (const float*)d_in[8];
  const float* bb   = (const float*)d_in[9];
  const float* We   = (const float*)d_in[10];
  const float* be   = (const float*)d_in[11];
  const float* Wa   = (const float*)d_in[12];
  const float* ba   = (const float*)d_in[13];
  const float* Wo   = (const float*)d_in[14];
  const float* bo   = (const float*)d_in[15];
  const float* mem0 = (const float*)d_in[16];
  float* out = (float*)d_out;

  // workspace carve (~10.05 MB total)
  char* p = (char*)d_ws;
  float*  Gin   = (float*)p;   p += (size_t)NS * 4096 * sizeof(float);
  __bf16* hr    = (__bf16*)p;  p += (size_t)NS * NKO * sizeof(__bf16);
  float*  memb  = (float*)p;   p += (size_t)2 * NMU * NMD * sizeof(float);
  float*  hbuf  = (float*)p;   p += (size_t)2 * NH * sizeof(float);
  float*  cbuf  = (float*)p;   p += (size_t)NH * sizeof(float);
  float*  readv = (float*)p;   p += (size_t)NMD * sizeof(float);
  float*  keyv  = (float*)p;   p += (size_t)NMD * sizeof(float);
  float*  erasv = (float*)p;   p += (size_t)NMD * sizeof(float);
  float*  addv  = (float*)p;   p += (size_t)NMD * sizeof(float);
  float*  betav = (float*)p;   p += (size_t)4 * sizeof(float);
  float*  simv  = (float*)p;   p += (size_t)NMU * sizeof(float);
  unsigned* syn = (unsigned*)p;

  ntm_init<<<dim3((NMU * NMD + 255) / 256), 256, 0, stream>>>(mem0, memb, hbuf, cbuf, readv, syn);
  // grid.x = M tiles (fast axis), grid.y = N tiles
  ntm_gin_gemm<<<dim3(NS / 32, 4096 / 256), 256, 0, stream>>>(x, emb, W_ih, b_ih, Gin);
  ntm_recur<<<dim3(NWGR), 256, 0, stream>>>(W_ih, W_hh, b_hh, Wk, bk, Wb, bb, We, be, Wa, ba,
                                            Gin, hr, memb, hbuf, cbuf, readv, keyv, erasv, addv,
                                            betav, simv, syn);
  // 4 M-blocks (fast axis) x 393 N-blocks; block tile 128x128
  ntm_out_gemm<<<dim3(NS / 128, (NV + 127) / 128), 256, 0, stream>>>(hr, Wo, bo, out);

  (void)in_sizes; (void)n_in; (void)out_size; (void)ws_size;
}
